// LSTM_56985626083979
// MI455X (gfx1250) — compile-verified, matrix operand order checked by reading, not run on hardware
//
#include <hip/hip_runtime.h>
#include <hip/hip_bf16.h>

typedef __attribute__((ext_vector_type(16))) __bf16 v16bf;
typedef __attribute__((ext_vector_type(8)))  float  v8f;

union Frag {
    v16bf v;
    uint4 q[2];
};

#define B_SZ   256
#define T_SZ   128
#define D_SZ   256
#define H_SZ   1024
#define KCAT   1280     // H + D
#define NCLS   1000
#define NPAD   1024
#define LDS_STRIDE 264  // 256 + 8 bf16 pad -> conflict-free b128 reads

// ---------------- one-time prep kernels ----------------

__global__ void k_f32_to_bf16(const float* __restrict__ src, __bf16* __restrict__ dst, int n) {
    int i = blockIdx.x * blockDim.x + threadIdx.x;
    if (i < n) dst[i] = (__bf16)src[i];
}

// Pre-swizzle the 4 gate weights (KCAT x H, row-major fp32) into per-lane WMMA
// B-fragment layout: block(j,q) of 32 lanes x 16 bf16, lane-contiguous 32B.
__global__ void k_repack_gates(const float* __restrict__ Wf, const float* __restrict__ Wi,
                               const float* __restrict__ Wc, const float* __restrict__ Wo,
                               __bf16* __restrict__ out) {
    const int PER_GATE = 64 * 40 * 512;             // (H/16) * (KCAT/32) * 32*16
    int idx = blockIdx.x * blockDim.x + threadIdx.x;
    if (idx >= 4 * PER_GATE) return;
    int g = idx / PER_GATE;
    int r = idx % PER_GATE;
    int i    = r & 15;
    int L    = (r >> 4) & 31;
    int rest = r >> 9;
    int q    = rest % 40;
    int j    = rest / 40;
    int half = L >> 4;
    int kl   = (i < 8) ? (8 * half + i) : (16 + 8 * half + (i - 8));
    int k    = q * 32 + kl;
    int n    = j * 16 + (L & 15);
    const float* W = (g == 0) ? Wf : (g == 1) ? Wi : (g == 2) ? Wc : Wo;
    out[idx] = (__bf16)W[(size_t)k * H_SZ + n];
}

// Pre-swizzle W_p (H x NCLS) into same B layout, N zero-padded to NPAD.
__global__ void k_repack_wp(const float* __restrict__ Wp, __bf16* __restrict__ out) {
    const int TOTAL = 64 * 32 * 512;                // (NPAD/16) * (H/32) * 32*16
    int idx = blockIdx.x * blockDim.x + threadIdx.x;
    if (idx >= TOTAL) return;
    int i    = idx & 15;
    int L    = (idx >> 4) & 31;
    int rest = idx >> 9;
    int q    = rest % 32;
    int j    = rest / 32;
    int half = L >> 4;
    int kl   = (i < 8) ? (8 * half + i) : (16 + 8 * half + (i - 8));
    int k    = q * 32 + kl;
    int n    = j * 16 + (L & 15);
    float v  = (n < NCLS) ? Wp[(size_t)k * NCLS + n] : 0.f;
    out[idx] = (__bf16)v;
}

__global__ void k_init_state(__bf16* __restrict__ h0, float* __restrict__ C, int n) {
    int i = blockIdx.x * blockDim.x + threadIdx.x;
    if (i < n) { h0[i] = (__bf16)0.f; C[i] = 0.f; }
}

// ---------------- recurrent step ----------------
// grid: (B/16, H/16); block: 128 threads = 4 waves, wave g handles gate g
// (0=f, 1=i, 2=C~, 3=o). K = 1280 over [h | x_t], 5 chunks of 256.

__launch_bounds__(128)
__global__ void k_lstm_step(const __bf16* __restrict__ hprev,
                            const __bf16* __restrict__ xbf, int t,
                            const __bf16* __restrict__ wswz,
                            const float* __restrict__ bfv, const float* __restrict__ biv,
                            const float* __restrict__ bcv,
                            float* __restrict__ Cst, __bf16* __restrict__ hnew) {
    __shared__ __attribute__((aligned(16))) __bf16 As[16 * LDS_STRIDE];
    __shared__ float Es[4][8][32];

    const int bm   = blockIdx.x;
    const int j    = blockIdx.y;
    const int tid  = threadIdx.x;
    const int g    = tid >> 5;           // gate / wave id
    const int lane = tid & 31;
    const int m16  = lane & 15;
    const int half = lane >> 4;

    v8f acc = {0.f, 0.f, 0.f, 0.f, 0.f, 0.f, 0.f, 0.f};
    const __bf16* wg = wswz + (size_t)(g * 64 + j) * 40 * 512;

    for (int c = 0; c < 5; ++c) {
        // cooperative, coalesced load of A chunk [16 rows x 256 K] into LDS
        #pragma unroll
        for (int it = 0; it < 4; ++it) {
            int lin  = it * 128 + tid;   // 0..511
            int row  = lin >> 5;
            int col8 = lin & 31;         // units of 8 bf16
            const __bf16* src;
            if (c < 4)
                src = hprev + (size_t)(bm * 16 + row) * H_SZ + c * 256 + col8 * 8;
            else
                src = xbf + ((size_t)(bm * 16 + row) * T_SZ + t) * D_SZ + col8 * 8;
            *reinterpret_cast<uint4*>(&As[row * LDS_STRIDE + col8 * 8]) =
                *reinterpret_cast<const uint4*>(src);
        }
        __syncthreads();

        if (c < 4)  // warm L2->L0 path for next chunk's B stream
            __builtin_prefetch(wg + (size_t)((c + 1) * 8) * 512 + lane * 16, 0, 1);

        #pragma unroll
        for (int q = 0; q < 8; ++q) {
            Frag a, b;
            const __bf16* ap = &As[m16 * LDS_STRIDE + q * 32 + 8 * half];
            a.q[0] = *reinterpret_cast<const uint4*>(ap);
            a.q[1] = *reinterpret_cast<const uint4*>(ap + 16);
            const __bf16* bp = wg + (size_t)(c * 8 + q) * 512 + lane * 16;
            b.q[0] = *reinterpret_cast<const uint4*>(bp);
            b.q[1] = *reinterpret_cast<const uint4*>(bp + 8);
            acc = __builtin_amdgcn_wmma_f32_16x16x32_bf16(
                false, a.v, false, b.v, (short)0, acc, false, false);
        }
        __syncthreads();
    }

    // fused activation epilogue
    const int n = j * 16 + m16;
    #pragma unroll
    for (int r = 0; r < 8; ++r) {
        float v = acc[r];
        float a;
        if (g == 0)      a = 1.f / (1.f + __expf(-(v + bfv[n])));
        else if (g == 1) a = 1.f / (1.f + __expf(-(v + biv[n])));
        else if (g == 2) a = tanhf(v + bcv[n]);
        else             a = 1.f / (1.f + __expf(-v));   // o-gate: no bias
        Es[g][r][lane] = a;
    }
    __syncthreads();

    if (g == 0) {
        #pragma unroll
        for (int r = 0; r < 8; ++r) {
            int m = bm * 16 + r + 8 * half;
            size_t idx = (size_t)m * H_SZ + n;
            float f  = Es[0][r][lane];
            float ig = Es[1][r][lane];
            float ct = Es[2][r][lane];
            float o  = Es[3][r][lane];
            float Cn = f * Cst[idx] + ig * ct;
            Cst[idx] = Cn;
            hnew[idx] = (__bf16)(o * tanhf(Cn));
        }
    }
}

// ---------------- classifier: h_final @ W_p + b_p ----------------
// grid: (B/16, NPAD/64); wave w handles n-tile blockIdx.y*4+w, K = 1024.

__launch_bounds__(128)
__global__ void k_classifier(const __bf16* __restrict__ h,
                             const __bf16* __restrict__ wp,
                             const float* __restrict__ bp,
                             float* __restrict__ out) {
    __shared__ __attribute__((aligned(16))) __bf16 As[16 * LDS_STRIDE];

    const int bm   = blockIdx.x;
    const int tid  = threadIdx.x;
    const int w    = tid >> 5;
    const int lane = tid & 31;
    const int j    = blockIdx.y * 4 + w;
    const int m16  = lane & 15;
    const int half = lane >> 4;

    v8f acc = {0.f, 0.f, 0.f, 0.f, 0.f, 0.f, 0.f, 0.f};
    const __bf16* wj = wp + (size_t)j * 32 * 512;

    for (int c = 0; c < 4; ++c) {
        #pragma unroll
        for (int it = 0; it < 4; ++it) {
            int lin  = it * 128 + tid;
            int row  = lin >> 5;
            int col8 = lin & 31;
            *reinterpret_cast<uint4*>(&As[row * LDS_STRIDE + col8 * 8]) =
                *reinterpret_cast<const uint4*>(
                    h + (size_t)(bm * 16 + row) * H_SZ + c * 256 + col8 * 8);
        }
        __syncthreads();
        #pragma unroll
        for (int q = 0; q < 8; ++q) {
            Frag a, b;
            const __bf16* ap = &As[m16 * LDS_STRIDE + q * 32 + 8 * half];
            a.q[0] = *reinterpret_cast<const uint4*>(ap);
            a.q[1] = *reinterpret_cast<const uint4*>(ap + 16);
            const __bf16* bpp = wj + (size_t)(c * 8 + q) * 512 + lane * 16;
            b.q[0] = *reinterpret_cast<const uint4*>(bpp);
            b.q[1] = *reinterpret_cast<const uint4*>(bpp + 8);
            acc = __builtin_amdgcn_wmma_f32_16x16x32_bf16(
                false, a.v, false, b.v, (short)0, acc, false, false);
        }
        __syncthreads();
    }

    const int n = j * 16 + m16;
    if (n < NCLS) {
        #pragma unroll
        for (int r = 0; r < 8; ++r) {
            int m = bm * 16 + r + 8 * half;
            out[(size_t)m * NCLS + n] = acc[r] + bp[n];
        }
    }
}

// ---------------- launch ----------------

extern "C" void kernel_launch(void* const* d_in, const int* in_sizes, int n_in,
                              void* d_out, int out_size, void* d_ws, size_t ws_size,
                              hipStream_t stream) {
    const float* x   = (const float*)d_in[0];
    const float* Wf  = (const float*)d_in[1];
    const float* bf_ = (const float*)d_in[2];
    const float* Wi  = (const float*)d_in[3];
    const float* bi_ = (const float*)d_in[4];
    const float* Wc  = (const float*)d_in[5];
    const float* bc_ = (const float*)d_in[6];
    const float* Wo  = (const float*)d_in[7];
    const float* Wp  = (const float*)d_in[8];
    const float* bp_ = (const float*)d_in[9];

    char* ws = (char*)d_ws;
    __bf16* xbf   = (__bf16*)(ws + 0);                 // 16,777,216 B
    __bf16* wswz  = (__bf16*)(ws + 16777216);          // 10,485,760 B
    __bf16* wpswz = (__bf16*)(ws + 27262976);          //  2,097,152 B
    __bf16* h0    = (__bf16*)(ws + 29360128);          //    524,288 B
    __bf16* h1    = (__bf16*)(ws + 29884416);          //    524,288 B
    float*  Cst   = (float*) (ws + 30408704);          //  1,048,576 B  (total ~31.5 MB)

    const int NX = B_SZ * T_SZ * D_SZ;                 // 8,388,608
    k_f32_to_bf16<<<(NX + 255) / 256, 256, 0, stream>>>(x, xbf, NX);

    const int NW = 4 * 64 * 40 * 512;                  // 5,242,880
    k_repack_gates<<<(NW + 255) / 256, 256, 0, stream>>>(Wf, Wi, Wc, Wo, wswz);

    const int NWP = 64 * 32 * 512;                     // 1,048,576
    k_repack_wp<<<(NWP + 255) / 256, 256, 0, stream>>>(Wp, wpswz);

    const int NS = B_SZ * H_SZ;                        // 262,144
    k_init_state<<<(NS + 255) / 256, 256, 0, stream>>>(h0, Cst, NS);

    for (int t = 0; t < T_SZ; ++t) {
        const __bf16* hp = (t & 1) ? h1 : h0;
        __bf16*       hn = (t & 1) ? h0 : h1;
        k_lstm_step<<<dim3(B_SZ / 16, H_SZ / 16), 128, 0, stream>>>(
            hp, xbf, t, wswz, bf_, bi_, bc_, Cst, hn);
    }
    // 128 steps: final h lands in h0
    k_classifier<<<dim3(B_SZ / 16, NPAD / 64), 128, 0, stream>>>(h0, wpswz, bp_, (float*)d_out);
}
